// RelativeMultiHeadAttention_10350871183649
// MI455X (gfx1250) — matrix operand order repaired and checked by reading, not compiled, not run on hardware
//
#include <hip/hip_runtime.h>
#include <stdint.h>

// Problem constants: B=4, S=1024, D=1024, H=16, DH=64
constexpr int cBn = 4;
constexpr int cS  = 1024;
constexpr int cD  = 1024;
constexpr int cH  = 16;
constexpr int cDH = 64;
constexpr float NEGV = -10000.0f;

// Flip to 0 if the async-to-LDS asm mnemonics are rejected by the assembler.
#ifndef USE_ASYNC_LDS
#define USE_ASYNC_LDS 1
#endif

typedef __attribute__((ext_vector_type(16))) __bf16 v16bf;
typedef __attribute__((ext_vector_type(8)))  float  v8f;
typedef __attribute__((ext_vector_type(8)))  __bf16 bf16x8;

// ---------------------------------------------------------------------------
// f32 -> bf16 conversion
// ---------------------------------------------------------------------------
__global__ __launch_bounds__(256)
void cvt_f32_to_bf16(const float* __restrict__ src, __bf16* __restrict__ dst, long n) {
    long i = (long)blockIdx.x * 256 + threadIdx.x;
    if (i < n) dst[i] = (__bf16)src[i];
}

// qu = bf16(q + u_bias[h,d]), qv = bf16(q + v_bias[h,d]); bias flat index = i % D
__global__ __launch_bounds__(256)
void head_bias_kernel(const float* __restrict__ q, const float* __restrict__ u,
                      const float* __restrict__ v, __bf16* __restrict__ qu,
                      __bf16* __restrict__ qv, long n) {
    long i = (long)blockIdx.x * 256 + threadIdx.x;
    if (i < n) {
        int c = (int)(i % cD);
        float qf = q[i];
        qu[i] = (__bf16)(qf + u[c]);
        qv[i] = (__bf16)(qf + v[c]);
    }
}

// vT[b,h,d,j] = v[b,j,h,d]
__global__ __launch_bounds__(256)
void vtrans_kernel(const __bf16* __restrict__ vb, __bf16* __restrict__ vt) {
    int j  = blockIdx.x * 256 + threadIdx.x;   // 0..S-1
    int d  = blockIdx.y;                       // 0..DH-1
    int bh = blockIdx.z;                       // b*H + h
    int b = bh / cH, h = bh % cH;
    vt[((long)bh * cDH + d) * cS + j] =
        vb[((long)b * cS + j) * cD + h * cDH + d];
}

// L[b] = S - sum_i mask[b, i, 0]
__global__ __launch_bounds__(256)
void len_kernel(const unsigned char* __restrict__ mask, int* __restrict__ L) {
    __shared__ int s[cBn];
    int t = threadIdx.x;
    if (t < cBn) s[t] = 0;
    __syncthreads();
    int b = t >> 6, i0 = t & 63;
    int acc = 0;
    for (int i = i0; i < cS; i += 64)
        acc += mask[(long)b * cS * cS + (long)i * cS];
    atomicAdd(&s[b], acc);
    __syncthreads();
    if (t < cBn) L[t] = cS - s[t];
}

// score = mask ? NEG : (content + rel_shift(pos)) / sqrt(D)   (in place on content)
__global__ __launch_bounds__(256)
void shift_combine_kernel(float* __restrict__ content, const float* __restrict__ pos,
                          const unsigned char* __restrict__ mask,
                          const int* __restrict__ Lbuf, float inv_sqrt_d) {
    int j  = blockIdx.x * 256 + threadIdx.x;
    int i  = blockIdx.y;
    int bh = blockIdx.z;
    int b  = bh / cH;
    long sidx = ((long)bh * cS + i) * cS + j;
    unsigned char mb = mask[((long)b * cS + i) * cS + j];
    float outv;
    if (mb) {
        outv = NEGV;
    } else {
        float shifted = 0.0f;
        int L = Lbuf[b];
        if (i < L && j < L) {
            int m   = L + i * L + j;
            int row = m / (L + 1);
            int col = m - row * (L + 1);
            if (col != 0) {
                int rr = min(row, cS - 1);
                int cc = min(max(col - 1, 0), cS - 1);
                shifted = pos[((long)bh * cS + rr) * cS + cc];
            }
        }
        outv = (content[sidx] + shifted) * inv_sqrt_d;
    }
    content[sidx] = outv;
}

// row softmax over j; write bf16 attn with masked positions zeroed
__global__ __launch_bounds__(256)
void softmax_kernel(const float* __restrict__ score, const unsigned char* __restrict__ mask,
                    __bf16* __restrict__ attn) {
    __shared__ float red[256];
    const int t  = threadIdx.x;
    const int i  = blockIdx.x;
    const int bh = blockIdx.y;
    const int b  = bh / cH;
    const float* row = score + ((long)bh * cS + i) * cS;
    const unsigned char* mrow = mask + ((long)b * cS + i) * cS;

    float vals[4];
    float mx = -3.4e38f;
#pragma unroll
    for (int q = 0; q < 4; ++q) { vals[q] = row[t * 4 + q]; mx = fmaxf(mx, vals[q]); }
    red[t] = mx; __syncthreads();
    for (int s = 128; s > 0; s >>= 1) { if (t < s) red[t] = fmaxf(red[t], red[t + s]); __syncthreads(); }
    mx = red[0]; __syncthreads();

    float sum = 0.0f;
#pragma unroll
    for (int q = 0; q < 4; ++q) { vals[q] = __expf(vals[q] - mx); sum += vals[q]; }
    red[t] = sum; __syncthreads();
    for (int s = 128; s > 0; s >>= 1) { if (t < s) red[t] += red[t + s]; __syncthreads(); }
    float inv = 1.0f / red[0];

    __bf16* orow = attn + ((long)bh * cS + i) * cS;
#pragma unroll
    for (int q = 0; q < 4; ++q) {
        int j = t * 4 + q;
        orow[j] = mrow[j] ? (__bf16)0.0f : (__bf16)(vals[q] * inv);
    }
}

// Generic LDS-offset extraction: on AMDGCN, a generic pointer to LDS carries the
// wave-relative LDS byte offset in its low 32 bits (flat aperture mapping).
__device__ __forceinline__ uint32_t lds_off32(const void* p) {
    return (uint32_t)(uintptr_t)p;
}

// ---------------------------------------------------------------------------
// Generic WMMA bf16 GEMM:  C[m,n] = sum_k A[m,k] * Bt[n,k]  (+ bias[n])
//   A row-major (lda), Bt N-major with contiguous K (ldb).
//   Batched over grid.z: z -> (zb = z/Hdim, zh = z%Hdim), per-operand strides.
//   Block = 256 threads = 8 waves (4x2); block tile 128(M) x 64(N) x 64(K).
//   Each wave owns a 32x32 output tile = 2x2 accumulators; 8 WMMA per K-tile.
//   Double-buffered LDS fed by async global->LDS copies (ASYNCcnt-tracked).
// ---------------------------------------------------------------------------
template<bool HAS_BIAS, bool OUT_F32, bool OUT_BF16>
__global__ __launch_bounds__(256)
void wmma_gemm_nt(const __bf16* __restrict__ A, const __bf16* __restrict__ Bt,
                  float* __restrict__ C, __bf16* __restrict__ Cb,
                  const float* __restrict__ bias,
                  int K, int lda, int ldb, int ldc, int ldcb,
                  long sAb, long sAh, long sBb, long sBh, long sCb, long sCh,
                  int Hdim) {
    constexpr int BM = 128, BN = 64, BK = 64, LDP = BK + 8;  // +8 pad vs bank conflicts
    __shared__ __align__(16) __bf16 As[2][BM][LDP];
    __shared__ __align__(16) __bf16 Bs[2][BN][LDP];

    const int tid  = threadIdx.x;
    const int lane = tid & 31;
    const int w    = tid >> 5;
    const int wm   = w >> 1;        // 0..3  (M sub-block of 32)
    const int wn   = w & 1;         // 0..1  (N sub-block of 32)
    const int m0   = blockIdx.y * BM;
    const int n0   = blockIdx.x * BN;
    const int z    = blockIdx.z;
    const int zb   = z / Hdim, zh = z % Hdim;

    A  += (long)zb * sAb + (long)zh * sAh;
    Bt += (long)zb * sBb + (long)zh * sBh;
    const long cbase = (long)zb * sCb + (long)zh * sCh;

    // staging coords: A tile 128x64 -> 32 elems/thread; B tile 64x64 -> 16 elems/thread
    const int ar = (tid * 32) >> 6, ac = (tid * 32) & 63;
    const int br = (tid * 16) >> 6, bc = (tid * 16) & 63;

    const int hi = lane >> 4;       // lane half (K sub-range / M offset)
    const int ln = lane & 15;

    v8f acc[2][2];
    acc[0][0] = {}; acc[0][1] = {}; acc[1][0] = {}; acc[1][1] = {};

    auto issue_tile = [&](int buf, int kt) {
        const __bf16* ga = A  + (long)(m0 + ar) * lda + kt + ac;
        const __bf16* gb = Bt + (long)(n0 + br) * ldb + kt + bc;
#if USE_ASYNC_LDS
        uint32_t la = lds_off32(&As[buf][ar][ac]);
        uint32_t lb = lds_off32(&Bs[buf][br][bc]);
#pragma unroll
        for (int c = 0; c < 4; ++c)
            asm volatile("global_load_async_to_lds_b128 %0, %1, off"
                         :: "v"(la + 16u * c), "v"(ga + 8 * c) : "memory");
#pragma unroll
        for (int c = 0; c < 2; ++c)
            asm volatile("global_load_async_to_lds_b128 %0, %1, off"
                         :: "v"(lb + 16u * c), "v"(gb + 8 * c) : "memory");
#else
#pragma unroll
        for (int c = 0; c < 4; ++c)
            *(bf16x8*)(&As[buf][ar][ac + 8 * c]) = *(const bf16x8*)(ga + 8 * c);
#pragma unroll
        for (int c = 0; c < 2; ++c)
            *(bf16x8*)(&Bs[buf][br][bc + 8 * c]) = *(const bf16x8*)(gb + 8 * c);
#endif
    };

    issue_tile(0, 0);
    int buf = 0;
    for (int kt = 0; kt < K; kt += BK) {
#if USE_ASYNC_LDS
        asm volatile("s_wait_asynccnt 0x0" ::: "memory");
#endif
        __syncthreads();                         // tile `buf` fully in LDS for all waves
        if (kt + BK < K) issue_tile(buf ^ 1, kt + BK);   // prefetch next into other buffer

#pragma unroll
        for (int ks = 0; ks < BK; ks += 32) {
            // Fragments in contiguous k order: element e of lane half h holds k = 16h + e (A and B alike).
            v16bf af[2], bg[2];
#pragma unroll
            for (int mi = 0; mi < 2; ++mi) {
                const int arow = wm * 32 + mi * 16 + ln;   // A: M = lane%16
#pragma unroll
                for (int t = 0; t < 8; ++t) {
                    int e  = 2 * t;
                    int ka = ks + e + hi * 16;
                    af[mi][e]     = As[buf][arow][ka];
                    af[mi][e + 1] = As[buf][arow][ka + 1];
                }
            }
#pragma unroll
            for (int ni = 0; ni < 2; ++ni) {
                const int bn = wn * 32 + ni * 16 + ln;     // B: N = lane%16
#pragma unroll
                for (int t = 0; t < 8; ++t) {
                    int e  = 2 * t;
                    int kb = ks + e + hi * 16;
                    bg[ni][e]     = Bs[buf][bn][kb];
                    bg[ni][e + 1] = Bs[buf][bn][kb + 1];
                }
            }
#pragma unroll
            for (int mi = 0; mi < 2; ++mi)
#pragma unroll
                for (int ni = 0; ni < 2; ++ni)
                    acc[mi][ni] = __builtin_amdgcn_wmma_f32_16x16x32_bf16(
                        false, af[mi], false, bg[ni], (short)0, acc[mi][ni], false, false);
        }
        buf ^= 1;
    }

    // C/D layout: lane%16 = N, vgpr d + 8*(lane/16) = M
#pragma unroll
    for (int mi = 0; mi < 2; ++mi)
#pragma unroll
        for (int ni = 0; ni < 2; ++ni) {
            int gn = n0 + wn * 32 + ni * 16 + ln;
            float bval = 0.0f;
            if constexpr (HAS_BIAS) bval = bias[gn];
#pragma unroll
            for (int d = 0; d < 8; ++d) {
                int gm = m0 + wm * 32 + mi * 16 + hi * 8 + d;
                float v = acc[mi][ni][d] + bval;
                if constexpr (OUT_F32)  C [cbase + (long)gm * ldc  + gn] = v;
                if constexpr (OUT_BF16) Cb[cbase + (long)gm * ldcb + gn] = (__bf16)v;
            }
        }
}

// ---------------------------------------------------------------------------
// Host-side orchestration
// ---------------------------------------------------------------------------
extern "C" void kernel_launch(void* const* d_in, const int* in_sizes, int n_in,
                              void* d_out, int out_size, void* d_ws, size_t ws_size,
                              hipStream_t stream) {
    (void)in_sizes; (void)n_in; (void)out_size; (void)ws_size;

    const float* query  = (const float*)d_in[0];
    const float* key    = (const float*)d_in[1];
    const float* value  = (const float*)d_in[2];
    const float* pos_e  = (const float*)d_in[3];
    const float* Wq     = (const float*)d_in[4];
    const float* bq     = (const float*)d_in[5];
    const float* Wk     = (const float*)d_in[6];
    const float* bk     = (const float*)d_in[7];
    const float* Wv     = (const float*)d_in[8];
    const float* bv     = (const float*)d_in[9];
    const float* Wpos   = (const float*)d_in[10];
    const float* u_bias = (const float*)d_in[11];
    const float* v_bias = (const float*)d_in[12];
    const float* Wout   = (const float*)d_in[13];
    const float* bout   = (const float*)d_in[14];
    const unsigned char* mask = (const unsigned char*)d_in[15];  // jnp bool -> 1 byte
    float* out = (float*)d_out;

    const long MD  = (long)cBn * cS * cD;       // 4M  activation elems
    const long DDn = (long)cD * cD;             // 1M  weight elems
    const long SCN = (long)cBn * cH * cS * cS;  // 64M score elems

    char* wp = (char*)d_ws;
    auto take = [&](long bytes) -> char* {
        char* r = wp; wp += (bytes + 255) & ~255L; return r;
    };
    __bf16* xq   = (__bf16*)take(MD * 2);
    __bf16* xk   = (__bf16*)take(MD * 2);
    __bf16* xv   = (__bf16*)take(MD * 2);
    __bf16* xp   = (__bf16*)take(MD * 2);
    __bf16* wqb  = (__bf16*)take(DDn * 2);
    __bf16* wkb  = (__bf16*)take(DDn * 2);
    __bf16* wvb  = (__bf16*)take(DDn * 2);
    __bf16* wpb  = (__bf16*)take(DDn * 2);
    __bf16* wob  = (__bf16*)take(DDn * 2);
    float*  qf   = (float*)take(MD * 4);
    __bf16* qu   = (__bf16*)take(MD * 2);
    __bf16* qv   = (__bf16*)take(MD * 2);
    __bf16* kb   = (__bf16*)take(MD * 2);
    __bf16* vb   = (__bf16*)take(MD * 2);
    __bf16* pb   = (__bf16*)take(MD * 2);
    __bf16* vt   = (__bf16*)take(MD * 2);
    float*  cont = (float*)take(SCN * 4);       // content scores -> combined scores
    float*  poss = (float*)take(SCN * 4);       // positional scores
    __bf16* attn = (__bf16*)take(SCN * 2);
    __bf16* ctxb = (__bf16*)take(MD * 2);
    int*    Lbuf = (int*)take(cBn * 4);

    const dim3 blk(256);
    auto cvt = [&](const float* s, __bf16* d, long n) {
        cvt_f32_to_bf16<<<dim3((unsigned)((n + 255) / 256)), blk, 0, stream>>>(s, d, n);
    };
    cvt(query, xq, MD);  cvt(key, xk, MD);  cvt(value, xv, MD);  cvt(pos_e, xp, MD);
    cvt(Wq, wqb, DDn);   cvt(Wk, wkb, DDn); cvt(Wv, wvb, DDn);
    cvt(Wpos, wpb, DDn); cvt(Wout, wob, DDn);

    // Projections: M=4096, N=1024, K=1024  (grid 16 x 32)
    dim3 gproj(cD / 64, (cBn * cS) / 128, 1);
    wmma_gemm_nt<true, true, false><<<gproj, blk, 0, stream>>>(
        xq, wqb, qf, nullptr, bq, cD, cD, cD, cD, 0, 0, 0, 0, 0, 0, 0, 1);
    wmma_gemm_nt<true, false, true><<<gproj, blk, 0, stream>>>(
        xk, wkb, nullptr, kb, bk, cD, cD, cD, 0, cD, 0, 0, 0, 0, 0, 0, 1);
    wmma_gemm_nt<true, false, true><<<gproj, blk, 0, stream>>>(
        xv, wvb, nullptr, vb, bv, cD, cD, cD, 0, cD, 0, 0, 0, 0, 0, 0, 1);
    wmma_gemm_nt<false, false, true><<<gproj, blk, 0, stream>>>(
        xp, wpb, nullptr, pb, nullptr, cD, cD, cD, 0, cD, 0, 0, 0, 0, 0, 0, 1);

    // q + u_bias / q + v_bias  -> bf16
    head_bias_kernel<<<dim3((unsigned)(MD / 256)), blk, 0, stream>>>(
        qf, u_bias, v_bias, qu, qv, MD);

    // Batched score GEMMs over z = b*H + h: M=N=S=1024, K=DH=64
    dim3 gscore(cS / 64, cS / 128, cBn * cH);
    const long sQb = (long)cS * cD, sQh = cDH;            // per-head activation slices
    const long sSb = (long)cH * cS * cS, sSh = (long)cS * cS;
    wmma_gemm_nt<false, true, false><<<gscore, blk, 0, stream>>>(
        qu, kb, cont, nullptr, nullptr, cDH, cD, cD, cS, 0,
        sQb, sQh, sQb, sQh, sSb, sSh, cH);
    wmma_gemm_nt<false, true, false><<<gscore, blk, 0, stream>>>(
        qv, pb, poss, nullptr, nullptr, cDH, cD, cD, cS, 0,
        sQb, sQh, sQb, sQh, sSb, sSh, cH);

    // Sequence lengths, relative shift + mask + scale (in place on cont)
    len_kernel<<<dim3(1), blk, 0, stream>>>(mask, Lbuf);
    shift_combine_kernel<<<dim3(cS / 256, cS, cBn * cH), blk, 0, stream>>>(
        cont, poss, mask, Lbuf, 1.0f / 32.0f);   // 1/sqrt(1024)

    // Softmax rows -> masked bf16 attention
    softmax_kernel<<<dim3(cS, cBn * cH), blk, 0, stream>>>(cont, mask, attn);

    // vT[b,h,d,j] for N-major B operand of attn @ v
    vtrans_kernel<<<dim3(cS / 256, cDH, cBn * cH), blk, 0, stream>>>(vb, vt);

    // context: per (b,h)  M=S, N=DH=64, K=S  -> bf16 ctx[b,s,h,d]
    dim3 gctx(cDH / 64, cS / 128, cBn * cH);
    wmma_gemm_nt<false, false, true><<<gctx, blk, 0, stream>>>(
        attn, vt, nullptr, ctxb, nullptr, cS, cS, cS, 0, cD,
        sSb, sSh, (long)cH * cDH * cS, (long)cDH * cS, (long)cS * cD, (long)cDH, cH);

    // output projection: ctx @ Wout^T + bout -> f32 d_out
    dim3 gout(cD / 64, (cBn * cS) / 128, 1);
    wmma_gemm_nt<true, true, false><<<gout, blk, 0, stream>>>(
        ctxb, wob, out, nullptr, bout, cD, cD, cD, cD, 0, 0, 0, 0, 0, 0, 0, 1);
}
